// TransitionPolicy_687194768173
// MI455X (gfx1250) — matrix operand order repaired and checked by reading, not compiled, not run on hardware
//
#include <hip/hip_runtime.h>
#include <hip/hip_bf16.h>

// ---------------------------------------------------------------------------
// TransitionPolicy rollout for MI455X (gfx1250, wave32, WMMA).
// All GEMMs: bf16 inputs, f32 accumulate via v_wmma_f32_16x16x32_bf16.
// Each 3-layer MLP fused in one kernel; hidden activations staged in LDS.
// MODE / A_REP / PK are compile-time template params so the row->(src,action)
// and moveaxis permutation math lowers to magic-number multiplies, not
// runtime integer division.
// ---------------------------------------------------------------------------

#define ACT_A   15
#define DIM     256
#define BATCH   32
#define GAM     0.99f

typedef __attribute__((ext_vector_type(16))) __bf16 v16bf;
typedef __attribute__((ext_vector_type(8)))  float  v8f;

static __device__ __forceinline__ v8f zero_v8f() {
  v8f z;
#pragma unroll
  for (int i = 0; i < 8; ++i) z[i] = 0.0f;
  return z;
}

// A-fragment (16x32 bf16) from fp32 row-major rows.
// Lane l: m = l&15, h = l>>4. VGPR v holds K pair k0 = 2*(v&3) + 16*(v>>2) + 8h.
// => elements 0..7  = k[kc+8h .. kc+8h+7]
//    elements 8..15 = k[kc+16+8h .. kc+16+8h+7]
static __device__ __forceinline__ v16bf load_a_f32(const float* __restrict__ rowPtr,
                                                   int kc, int h) {
  const float* p0 = rowPtr + kc + 8 * h;
  const float* p1 = rowPtr + kc + 16 + 8 * h;
  v16bf r;
#pragma unroll
  for (int i = 0; i < 8; ++i) { r[i] = (__bf16)p0[i]; r[8 + i] = (__bf16)p1[i]; }
  return r;
}

static __device__ __forceinline__ v16bf load_a_bf16(const __bf16* __restrict__ rowPtr,
                                                    int kc, int h) {
  const __bf16* p0 = rowPtr + kc + 8 * h;
  const __bf16* p1 = rowPtr + kc + 16 + 8 * h;
  v16bf r;
#pragma unroll
  for (int i = 0; i < 8; ++i) { r[i] = p0[i]; r[8 + i] = p1[i]; }
  return r;
}

// B-fragment (32x16 bf16) from weights stored transposed [N][K] (K contiguous).
// Lane l: n = l&15, h = l>>4; elements 0..7 = k[kc+8h..+7], 8..15 = k[kc+16+8h..+7].
static __device__ __forceinline__ v16bf load_b_bf16(const __bf16* __restrict__ nRow,
                                                    int kc, int h) {
  const __bf16* p0 = nRow + kc + 8 * h;
  const __bf16* p1 = nRow + kc + 16 + 8 * h;
  v16bf r;
#pragma unroll
  for (int i = 0; i < 8; ++i) { r[i] = p0[i]; r[8 + i] = p1[i]; }
  return r;
}

static __device__ __forceinline__ v8f wmma_bf16(v16bf a, v16bf b, v8f c) {
  return __builtin_amdgcn_wmma_f32_16x16x32_bf16(false, a, false, b, (short)0, c,
                                                 false, false);
}

// Transpose + convert fp32 W[K x N] (row-major) -> bf16 Wt[N][K].
__global__ void transpose_cvt_kernel(const float* __restrict__ W,
                                     __bf16* __restrict__ Wt, int K, int N) {
  int idx = blockIdx.x * blockDim.x + threadIdx.x;
  if (idx >= K * N) return;
  int n = idx / K;
  int k = idx % K;
  Wt[n * K + k] = (__bf16)W[k * N + n];
}

// ---------------------------------------------------------------------------
// Fused 3-layer MLP over 32-row tiles.
//   layer1: relu(src @ W0 + b0 [+ a * wlast0])   (K=256, action feature = rank-1)
//   layer2: relu(h1 @ W1 + b1)
//   layer3: h2 @ W2 + b2  (width n2)
// MODE 0: transition -> write fp32 rows to out0 with moveaxis permutation (PK)
// MODE 1: cont/rew   -> out0[g] = done-logit, out1[g] = reward
// MODE 2: value      -> out0[row] = (layer3 . vW) + vb
// A_REP > 0: GEMM row g draws src row g/A_REP with action scalar g%A_REP.
// ---------------------------------------------------------------------------
#define LDS_STRIDE 264

template <int MODE, int A_REP, int PK>
__global__ __launch_bounds__(256) void fused_mlp3_kernel(
    const float* __restrict__ src,
    const __bf16* __restrict__ Wt0, const float* __restrict__ b0,
    const float* __restrict__ wlast0,
    const __bf16* __restrict__ Wt1, const float* __restrict__ b1,
    const __bf16* __restrict__ Wt2, const float* __restrict__ b2, int n2,
    float* __restrict__ out0, float* __restrict__ out1,
    const float* __restrict__ vW, const float* __restrict__ vb) {
  __shared__ __bf16 act0[32][LDS_STRIDE];
  __shared__ __bf16 act1[32][LDS_STRIDE];
  __shared__ float vsum[32];

  const int tid = threadIdx.x;
  const int lane = tid & 31;
  const int wave = tid >> 5;
  const int wr = wave >> 2;   // 0..1  row-tile of 16
  const int wc = wave & 3;    // 0..3  col-tile of 64
  const int h = lane >> 4;
  const int ln = lane & 15;
  const int rowTile = wr * 16;
  const int gRow0 = blockIdx.x * 32;

  if (MODE == 2) {
    if (tid < 32) vsum[tid] = 0.0f;
  }

  // Per-lane A-fragment source row (compile-time divisor -> magic multiply).
  const int gA = gRow0 + rowTile + ln;
  const int srcRow = (A_REP > 0) ? (gA / A_REP) : gA;
  const float* aPtr = src + (long)srcRow * DIM;

  v8f acc[4];

  // ---------------- layer 1 (A: fp32 global -> bf16) ----------------
#pragma unroll
  for (int t = 0; t < 4; ++t) acc[t] = zero_v8f();
  for (int kc = 0; kc < DIM; kc += 32) {
    if (kc + 32 < DIM) __builtin_prefetch(aPtr + kc + 32, 0, 1);
    v16bf af = load_a_f32(aPtr, kc, h);
#pragma unroll
    for (int t = 0; t < 4; ++t) {
      int n = wc * 64 + t * 16 + ln;
      v16bf bf = load_b_bf16(Wt0 + (long)n * DIM, kc, h);
      acc[t] = wmma_bf16(af, bf, acc[t]);
    }
  }
#pragma unroll
  for (int t = 0; t < 4; ++t) {
    int col = wc * 64 + t * 16 + ln;
    float bb = b0[col];
    float wl = (A_REP > 0) ? wlast0[col] : 0.0f;
#pragma unroll
    for (int i = 0; i < 8; ++i) {
      int m = i + 8 * h;
      int g = gRow0 + rowTile + m;
      float av = (A_REP > 0) ? (float)(g % A_REP) : 0.0f;
      float v = acc[t][i] + bb + av * wl;
      act0[rowTile + m][col] = (__bf16)fmaxf(v, 0.0f);
    }
  }
  __syncthreads();

  // ---------------- layer 2 (A: LDS bf16) ----------------
#pragma unroll
  for (int t = 0; t < 4; ++t) acc[t] = zero_v8f();
  {
    const __bf16* aRow = &act0[rowTile + ln][0];
    for (int kc = 0; kc < DIM; kc += 32) {
      v16bf af = load_a_bf16(aRow, kc, h);
#pragma unroll
      for (int t = 0; t < 4; ++t) {
        int n = wc * 64 + t * 16 + ln;
        v16bf bf = load_b_bf16(Wt1 + (long)n * DIM, kc, h);
        acc[t] = wmma_bf16(af, bf, acc[t]);
      }
    }
  }
#pragma unroll
  for (int t = 0; t < 4; ++t) {
    int col = wc * 64 + t * 16 + ln;
    float bb = b1[col];
#pragma unroll
    for (int i = 0; i < 8; ++i) {
      int m = i + 8 * h;
      act1[rowTile + m][col] = (__bf16)fmaxf(acc[t][i] + bb, 0.0f);
    }
  }
  __syncthreads();

  // ---------------- layer 3 ----------------
  if (MODE == 1) {
    // Only 2 output columns; waves with wc!=0 idle (uniform branch).
    if (wc == 0) {
      v8f c = zero_v8f();
      const __bf16* aRow = &act1[rowTile + ln][0];
      for (int kc = 0; kc < DIM; kc += 32) {
        v16bf af = load_a_bf16(aRow, kc, h);
        const __bf16* bp = Wt2 + (long)((ln < n2) ? ln : 0) * DIM;
        v16bf bf = load_b_bf16(bp, kc, h);
        if (ln >= n2) {
#pragma unroll
          for (int i = 0; i < 16; ++i) bf[i] = (__bf16)0.0f;
        }
        c = wmma_bf16(af, bf, c);
      }
      if (ln < 2) {
        float bb = b2[ln];
#pragma unroll
        for (int i = 0; i < 8; ++i) {
          int m = i + 8 * h;
          int g = gRow0 + rowTile + m;
          float v = c[i] + bb;
          if (ln == 0) out0[g] = v;  // done logit
          else         out1[g] = v;  // reward
        }
      }
    }
  } else {
#pragma unroll
    for (int t = 0; t < 4; ++t) acc[t] = zero_v8f();
    const __bf16* aRow = &act1[rowTile + ln][0];
    for (int kc = 0; kc < DIM; kc += 32) {
      v16bf af = load_a_bf16(aRow, kc, h);
#pragma unroll
      for (int t = 0; t < 4; ++t) {
        int n = wc * 64 + t * 16 + ln;
        v16bf bf = load_b_bf16(Wt2 + (long)n * DIM, kc, h);
        acc[t] = wmma_bf16(af, bf, acc[t]);
      }
    }
    if (MODE == 0) {
      // Transition: write fp32 rows with moveaxis(-2,1) fold:
      // g = src*A + a ; b = src/PK ; rest = src%PK ; out = (b*A + a)*PK + rest
#pragma unroll
      for (int t = 0; t < 4; ++t) {
        int col = wc * 64 + t * 16 + ln;
        float bb = b2[col];
#pragma unroll
        for (int i = 0; i < 8; ++i) {
          int m = i + 8 * h;
          int g = gRow0 + rowTile + m;
          int a = g % A_REP;
          int sr = g / A_REP;
          int bidx = sr / PK;
          int rest = sr % PK;
          long outRow = ((long)(bidx * A_REP + a)) * PK + rest;
          out0[outRow * DIM + col] = acc[t][i] + bb;
        }
      }
    } else {  // MODE == 2: value head (dot with vW, +vb)
#pragma unroll
      for (int t = 0; t < 4; ++t) {
        int col = wc * 64 + t * 16 + ln;
        float bb = b2[col];
        float w = vW[col];
#pragma unroll
        for (int i = 0; i < 8; ++i) {
          int m = i + 8 * h;
          atomicAdd(&vsum[rowTile + m], (acc[t][i] + bb) * w);
        }
      }
      __syncthreads();
      if (tid < 32) out0[gRow0 + tid] = vsum[tid] + vb[0];
    }
  }
}

// ---------------------------------------------------------------------------
// Final collapse: vs3[b,p,q,r] = g^3*val3 + g^2*rew0[b,r] + g*rew1[b,q,r],
// masked by cont logits (> 0 == round(sigmoid)==1), two softmax-weighted sums
// over 15-wide axes, then log_softmax -> log_probs [32,15].
// ---------------------------------------------------------------------------
__global__ __launch_bounds__(256) void finalize_kernel(
    const float* __restrict__ val3, const float* __restrict__ rew0,
    const float* __restrict__ rew1, const float* __restrict__ cont1,
    const float* __restrict__ cont2, float* __restrict__ logp) {
  int b = blockIdx.x;
  __shared__ float vs2[225];
  __shared__ float vs1[15];
  const float g1 = GAM, g2 = GAM * GAM, g3 = GAM * GAM * GAM;

  for (int pq = threadIdx.x; pq < 225; pq += blockDim.x) {
    int p = pq / 15, q = pq % 15;
    float vals[15];
    float mx = -1e30f;
#pragma unroll
    for (int r = 0; r < 15; ++r) {
      long idx = (((long)b * 15 + p) * 15 + q) * 15 + r;
      float v = g3 * val3[idx] + g2 * rew0[b * 15 + r] + g1 * rew1[(b * 15 + q) * 15 + r];
      if (cont2[idx] > 0.0f) v = 0.0f;
      vals[r] = v;
      mx = fmaxf(mx, v);
    }
    float se = 0.0f, sv = 0.0f;
#pragma unroll
    for (int r = 0; r < 15; ++r) {
      float e = __expf(vals[r] - mx);
      se += e;
      sv += e * vals[r];
    }
    vs2[pq] = sv / se;
  }
  __syncthreads();

  if (threadIdx.x < 15) {
    int p = threadIdx.x;
    float vals[15];
    float mx = -1e30f;
#pragma unroll
    for (int q = 0; q < 15; ++q) {
      float v = vs2[p * 15 + q];
      if (cont1[(b * 15 + p) * 15 + q] > 0.0f) v = 0.0f;
      vals[q] = v;
      mx = fmaxf(mx, v);
    }
    float se = 0.0f, sv = 0.0f;
#pragma unroll
    for (int q = 0; q < 15; ++q) {
      float e = __expf(vals[q] - mx);
      se += e;
      sv += e * vals[q];
    }
    vs1[p] = sv / se;
  }
  __syncthreads();

  if (threadIdx.x == 0) {
    float mx = -1e30f;
    for (int p = 0; p < 15; ++p) mx = fmaxf(mx, vs1[p]);
    float se = 0.0f;
    for (int p = 0; p < 15; ++p) se += __expf(vs1[p] - mx);
    float ls = __logf(se) + mx;
    for (int p = 0; p < 15; ++p) logp[b * 15 + p] = vs1[p] - ls;
  }
}

// ---------------------------------------------------------------------------
extern "C" void kernel_launch(void* const* d_in, const int* in_sizes, int n_in,
                              void* d_out, int out_size, void* d_ws, size_t ws_size,
                              hipStream_t stream) {
  (void)in_sizes; (void)n_in; (void)out_size; (void)ws_size;
  const float* x     = (const float*)d_in[0];
  const float* embW0 = (const float*)d_in[1];
  const float* embb0 = (const float*)d_in[2];
  const float* embW1 = (const float*)d_in[3];
  const float* embb1 = (const float*)d_in[4];
  const float* embW2 = (const float*)d_in[5];
  const float* embb2 = (const float*)d_in[6];
  const float* vW    = (const float*)d_in[7];
  const float* vb    = (const float*)d_in[8];
  const float* crW0  = (const float*)d_in[9];
  const float* crb0  = (const float*)d_in[10];
  const float* crW1  = (const float*)d_in[11];
  const float* crb1  = (const float*)d_in[12];
  const float* crW2  = (const float*)d_in[13];
  const float* crb2  = (const float*)d_in[14];
  const float* tmW0  = (const float*)d_in[15];
  const float* tmb0  = (const float*)d_in[16];
  const float* tmW1  = (const float*)d_in[17];
  const float* tmb1  = (const float*)d_in[18];
  const float* tmW2  = (const float*)d_in[19];
  const float* tmb2  = (const float*)d_in[20];

  char* ws = (char*)d_ws;
  size_t off = 0;
  auto alloc = [&](size_t bytes) -> void* {
    void* p = ws + off;
    off = (off + bytes + 255) & ~(size_t)255;
    return p;
  };

  const size_t WB = 256 * 256 * sizeof(__bf16);
  __bf16* embT0 = (__bf16*)alloc(WB);
  __bf16* embT1 = (__bf16*)alloc(WB);
  __bf16* embT2 = (__bf16*)alloc(WB);
  __bf16* crT0  = (__bf16*)alloc(WB);
  __bf16* crT1  = (__bf16*)alloc(WB);
  __bf16* crT2  = (__bf16*)alloc(2 * 256 * sizeof(__bf16));
  __bf16* tmT0  = (__bf16*)alloc(WB);
  __bf16* tmT1  = (__bf16*)alloc(WB);
  __bf16* tmT2  = (__bf16*)alloc(WB);

  const int R1 = BATCH * ACT_A;              // 480
  const int R2 = R1 * ACT_A;                 // 7200
  const int R3 = R2 * ACT_A;                 // 108000
  float* s1    = (float*)alloc((size_t)R1 * DIM * 4);
  float* s2    = (float*)alloc((size_t)R2 * DIM * 4);
  float* s3    = (float*)alloc((size_t)R3 * DIM * 4);
  float* cont0 = (float*)alloc((size_t)R1 * 4);
  float* rew0  = (float*)alloc((size_t)R1 * 4);
  float* cont1 = (float*)alloc((size_t)R2 * 4);
  float* rew1  = (float*)alloc((size_t)R2 * 4);
  float* cont2 = (float*)alloc((size_t)R3 * 4);
  float* rew2  = (float*)alloc((size_t)R3 * 4);
  float* val3  = (float*)alloc((size_t)R3 * 4);

  auto tc = [&](const float* W, __bf16* Wt, int N) {
    int total = 256 * N;
    transpose_cvt_kernel<<<(total + 255) / 256, 256, 0, stream>>>(W, Wt, 256, N);
  };
  tc(embW0, embT0, 256); tc(embW1, embT1, 256); tc(embW2, embT2, 256);
  tc(crW0, crT0, 256);   tc(crW1, crT1, 256);   tc(crW2, crT2, 2);
  tc(tmW0, tmT0, 256);   tc(tmW1, tmT1, 256);   tc(tmW2, tmT2, 256);

  float* logp = (float*)d_out;
  float* vout = (float*)d_out + BATCH * ACT_A;  // v after log_probs
  const float* crLast = crW0 + 256 * 256;       // 257th input row (action weights)
  const float* tmLast = tmW0 + 256 * 256;

  // v = value(x)
  fused_mlp3_kernel<2, 0, 1><<<1, 256, 0, stream>>>(
      x, embT0, embb0, nullptr, embT1, embb1, embT2, embb2, 256,
      vout, nullptr, vW, vb);
  // step 0 (480 rows)
  fused_mlp3_kernel<1, ACT_A, 1><<<R1 / 32, 256, 0, stream>>>(
      x, crT0, crb0, crLast, crT1, crb1, crT2, crb2, 2,
      cont0, rew0, nullptr, nullptr);
  fused_mlp3_kernel<0, ACT_A, 1><<<R1 / 32, 256, 0, stream>>>(
      x, tmT0, tmb0, tmLast, tmT1, tmb1, tmT2, tmb2, 256,
      s1, nullptr, nullptr, nullptr);
  // step 1 (7200 rows)
  fused_mlp3_kernel<1, ACT_A, 1><<<R2 / 32, 256, 0, stream>>>(
      s1, crT0, crb0, crLast, crT1, crb1, crT2, crb2, 2,
      cont1, rew1, nullptr, nullptr);
  fused_mlp3_kernel<0, ACT_A, ACT_A><<<R2 / 32, 256, 0, stream>>>(
      s1, tmT0, tmb0, tmLast, tmT1, tmb1, tmT2, tmb2, 256,
      s2, nullptr, nullptr, nullptr);
  // step 2 (108000 rows)
  fused_mlp3_kernel<1, ACT_A, 1><<<R3 / 32, 256, 0, stream>>>(
      s2, crT0, crb0, crLast, crT1, crb1, crT2, crb2, 2,
      cont2, rew2, nullptr, nullptr);
  fused_mlp3_kernel<0, ACT_A, ACT_A * ACT_A><<<R3 / 32, 256, 0, stream>>>(
      s2, tmT0, tmb0, tmLast, tmT1, tmb1, tmT2, tmb2, 256,
      s3, nullptr, nullptr, nullptr);
  // value(final states)
  fused_mlp3_kernel<2, 0, 1><<<R3 / 32, 256, 0, stream>>>(
      s3, embT0, embb0, nullptr, embT1, embb1, embT2, embb2, 256,
      val3, nullptr, vW, vb);
  // collapse to log_probs
  finalize_kernel<<<BATCH, 256, 0, stream>>>(val3, rew0, rew1, cont1, cont2, logp);
}
